// RGCNLayer_29764123361471
// MI455X (gfx1250) — matrix operand order repaired and checked by reading, not compiled, not run on hardware
//
#include <hip/hip_runtime.h>

// Problem constants (match reference)
#define N_NODES 50000
#define R_REL   16
#define H0      128
#define H1      64
#define NBASES  8
#define MTILES  ((N_NODES + 15) / 16)   // 3125

typedef __attribute__((ext_vector_type(16))) __bf16 v16bf;
typedef __attribute__((ext_vector_type(8)))  float  v8f;

// ---------------- init kernels ----------------
__global__ void init_counts_kernel(float* __restrict__ counts) {
    int i = blockIdx.x * blockDim.x + threadIdx.x;
    if (i < R_REL * N_NODES) counts[i] = 0.0f;
}

__global__ void init_out_kernel(float* __restrict__ out, const float* __restrict__ bias) {
    int i = blockIdx.x * blockDim.x + threadIdx.x;
    if (i < N_NODES * H1) out[i] = bias[i & (H1 - 1)];
}

// ---------------- edge counting: counts[rel*N + src] += 1 ----------------
__global__ void count_edges_kernel(const int* __restrict__ src, const int* __restrict__ rel,
                                   float* __restrict__ counts, int E) {
    int e = blockIdx.x * blockDim.x + threadIdx.x;
    if (e < E) atomicAdd(&counts[rel[e] * N_NODES + src[e]], 1.0f);
}

// ---------------- fp32 -> bf16 node features ----------------
__global__ void cvt_nodes_kernel(const float* __restrict__ nodes, __bf16* __restrict__ nb) {
    int i = blockIdx.x * blockDim.x + threadIdx.x;
    if (i < N_NODES * H0) nb[i] = (__bf16)nodes[i];
}

// ---------------- weights[r,i,o] = sum_b comps[r,b]*bases[b,i,o], stored transposed (r,o,i) bf16 ----------------
__global__ void make_weights_kernel(const float* __restrict__ comps, const float* __restrict__ bases,
                                    __bf16* __restrict__ wT) {
    int i = blockIdx.x * blockDim.x + threadIdx.x;   // flat (r, o, i0)
    if (i >= R_REL * H1 * H0) return;
    int i0 = i % H0;
    int o  = (i / H0) % H1;
    int r  = i / (H0 * H1);
    float acc = 0.0f;
#pragma unroll
    for (int b = 0; b < NBASES; ++b)
        acc += comps[r * NBASES + b] * bases[((size_t)b * H0 + i0) * H1 + o];
    wT[i] = (__bf16)acc;   // layout (r, o, i0) row-major == flat index i
}

// ---------------- batched GEMM via WMMA: nw[r, m, :] = nodes_bf16[m,:] . W[r] ----------------
// One wave computes a full 16x64 output strip (4 WMMA tiles sharing one A fragment per K-step).
// grid: (ceil(MTILES/4), R_REL), block: 128 threads = 4 waves, wave w -> m-tile blockIdx.x*4+w
__global__ void __launch_bounds__(128)
gemm_nw_kernel(const __bf16* __restrict__ nb, const __bf16* __restrict__ wT,
               float* __restrict__ nw) {
    const int wave  = threadIdx.x >> 5;
    const int mtile = blockIdx.x * 4 + wave;
    if (mtile >= MTILES) return;              // wave-uniform: EXEC stays all-ones
    const int r     = blockIdx.y;             // 0..15
    const int lane  = threadIdx.x & 31;
    const int half  = lane >> 4;              // lane half (K-split)
    const int lidx  = lane & 15;              // M row (A) / N col (B,D)
    const int mbase = mtile * 16;

    // A: row m = mbase+lidx of nodes (K contiguous along H0)
    const __bf16* rowA = nb + (size_t)(mbase + lidx) * H0;
    // B: column n = nt*16 + lidx of W[r] == row of transposed weights (K contiguous);
    // n-tile nt offset = nt*16*H0 elements
    const __bf16* rowB = wT + ((size_t)r * H1 + lidx) * H0;

    v8f c0 = {}, c1 = {}, c2 = {}, c3 = {};
#pragma unroll
    for (int ks = 0; ks < 4; ++ks) {          // K = 128 in 4 chunks of 32
        const int k0 = ks * 32;
        v16bf a;
#pragma unroll
        for (int j = 0; j < 8; ++j) {
            // A 16x32 bf16 layout: VGPR j<4: K = half*8 + 2j; VGPR j>=4: K = 16 + half*8 + 2(j-4)
            int ka = k0 + ((j >> 2) << 4) + (half << 3) + ((j & 3) << 1);
            a[2 * j]     = rowA[ka];
            a[2 * j + 1] = rowA[ka + 1];
        }
        v16bf b0, b1, b2, b3;
#pragma unroll
        for (int j = 0; j < 8; ++j) {
            // B 32x16 bf16 layout (matches A): VGPR j<4: K = half*8 + 2j; VGPR j>=4: K = 16 + half*8 + 2(j-4)
            int kb = k0 + ((j >> 2) << 4) + (half << 3) + ((j & 3) << 1);
            b0[2 * j] = rowB[kb];               b0[2 * j + 1] = rowB[kb + 1];
            b1[2 * j] = rowB[16 * H0 + kb];     b1[2 * j + 1] = rowB[16 * H0 + kb + 1];
            b2[2 * j] = rowB[32 * H0 + kb];     b2[2 * j + 1] = rowB[32 * H0 + kb + 1];
            b3[2 * j] = rowB[48 * H0 + kb];     b3[2 * j + 1] = rowB[48 * H0 + kb + 1];
        }
        c0 = __builtin_amdgcn_wmma_f32_16x16x32_bf16(false, a, false, b0, (short)0, c0, false, false);
        c1 = __builtin_amdgcn_wmma_f32_16x16x32_bf16(false, a, false, b1, (short)0, c1, false, false);
        c2 = __builtin_amdgcn_wmma_f32_16x16x32_bf16(false, a, false, b2, (short)0, c2, false, false);
        c3 = __builtin_amdgcn_wmma_f32_16x16x32_bf16(false, a, false, b3, (short)0, c3, false, false);
    }

    // D 16x16 f32 layout: VGPR j -> row M = j + 8*half, col = lidx (+ 16*nt)
    float* outRow = nw + (((size_t)r * N_NODES + mbase) * H1);
#pragma unroll
    for (int j = 0; j < 8; ++j) {
        size_t rowOff = (size_t)((half << 3) + j) * H1 + lidx;
        outRow[rowOff]      = c0[j];
        outRow[rowOff + 16] = c1[j];
        outRow[rowOff + 32] = c2[j];
        outRow[rowOff + 48] = c3[j];
    }
}

// ---------------- edge gather/scale/scatter: out[src] += nw[rel,dst] / counts[rel,src] ----------------
// 16 threads per edge, each handles one float4 (H1 = 64 = 16 * 4)
__global__ void scatter_edges_kernel(const int* __restrict__ src, const int* __restrict__ rel,
                                     const int* __restrict__ dst,
                                     const float* __restrict__ counts,
                                     const float4* __restrict__ nw4,
                                     float* __restrict__ out, int E) {
    int t = blockIdx.x * blockDim.x + threadIdx.x;
    int e = t >> 4;
    if (e >= E) return;
    int q  = t & 15;
    int s  = src[e];
    int rr = rel[e];
    int d  = dst[e];
    float val = 1.0f / counts[rr * N_NODES + s];
    float4 v = nw4[(((size_t)rr * N_NODES + d) * (H1 / 4)) + q];
    float* op = out + (size_t)s * H1 + q * 4;
    atomicAdd(op + 0, v.x * val);
    atomicAdd(op + 1, v.y * val);
    atomicAdd(op + 2, v.z * val);
    atomicAdd(op + 3, v.w * val);
}

extern "C" void kernel_launch(void* const* d_in, const int* in_sizes, int n_in,
                              void* d_out, int out_size, void* d_ws, size_t ws_size,
                              hipStream_t stream) {
    const float* nodes = (const float*)d_in[0];
    const float* comps = (const float*)d_in[1];
    const float* bases = (const float*)d_in[2];
    const float* bias  = (const float*)d_in[3];
    const int*   src   = (const int*)d_in[4];
    const int*   rel   = (const int*)d_in[5];
    const int*   dst   = (const int*)d_in[6];
    float* out = (float*)d_out;
    const int E = in_sizes[4];

    // workspace carve-up (256B aligned)
    char* ws = (char*)d_ws;
    size_t off = 0;
    auto carve = [&](size_t bytes) -> void* {
        void* p = ws + off;
        off += (bytes + 255) & ~(size_t)255;
        return p;
    };
    float*  counts = (float*) carve((size_t)R_REL * N_NODES * sizeof(float));      // 3.2 MB
    __bf16* nb     = (__bf16*)carve((size_t)N_NODES * H0 * sizeof(__bf16));        // 12.8 MB
    __bf16* wT     = (__bf16*)carve((size_t)R_REL * H1 * H0 * sizeof(__bf16));     // 256 KB
    float*  nw     = (float*) carve((size_t)R_REL * N_NODES * H1 * sizeof(float)); // 204.8 MB
    (void)ws_size;

    init_counts_kernel<<<(R_REL * N_NODES + 255) / 256, 256, 0, stream>>>(counts);
    init_out_kernel<<<(N_NODES * H1 + 255) / 256, 256, 0, stream>>>(out, bias);
    count_edges_kernel<<<(E + 255) / 256, 256, 0, stream>>>(src, rel, counts, E);
    cvt_nodes_kernel<<<(N_NODES * H0 + 255) / 256, 256, 0, stream>>>(nodes, nb);
    make_weights_kernel<<<(R_REL * H1 * H0 + 255) / 256, 256, 0, stream>>>(comps, bases, wT);
    gemm_nw_kernel<<<dim3((MTILES + 3) / 4, R_REL), 128, 0, stream>>>(nb, wT, nw);
    scatter_edges_kernel<<<((size_t)E * 16 + 255) / 256, 256, 0, stream>>>(src, rel, dst, counts,
                                                                           (const float4*)nw, out, E);
}
